// Decoder_80968723464309
// MI455X (gfx1250) — matrix operand order repaired and checked by reading, not compiled, not run on hardware
//
#include <hip/hip_runtime.h>
#include <hip/hip_bf16.h>

// ---------------------------------------------------------------------------
// Problem constants (from reference): B=32, T=64, S=512, H=1024, IN=512
// ---------------------------------------------------------------------------
#define BB 32
#define TT 64
#define SS 512
#define HH 1024
#define INW 512
#define GG (4 * HH)   // 4096 gates

typedef __attribute__((ext_vector_type(16))) __bf16 v16bf;
typedef __attribute__((ext_vector_type(8)))  float  v8f;

// ---------------------------------------------------------------------------
// WMMA fragment loaders (contiguous k order k = 16h + s used by both A and B, bf16 16x16x32)
// A (16x32, MxK) row-major, lda in elements:
//   lane<16 : row = lane,    elems = K(k..k+15)
//   lane>=16: row = lane-16, elems = K(k+16..k+31)
// B (32x16, KxN) given as row-major W[N][K] (so B[k][n] = W[n][k]):
//   lane<16 : col = lane,    elems = K(k..k+15)   (contiguous in W row)
//   lane>=16: col = lane-16, elems = K(k+16..k+31)
// ---------------------------------------------------------------------------
__device__ __forceinline__ v16bf load_a_frag(const __bf16* __restrict__ A, int lda,
                                             int rowBase, int kBase, int lane) {
  const int sel = lane >> 4;
  const int row = rowBase + (lane & 15);
  const __bf16* p = A + (size_t)row * lda + kBase + sel * 16;
  v16bf f;
  uint4* fp = reinterpret_cast<uint4*>(&f);
  fp[0] = *reinterpret_cast<const uint4*>(p);
  fp[1] = *reinterpret_cast<const uint4*>(p + 8);
  return f;
}

__device__ __forceinline__ v16bf load_b_frag(const __bf16* __restrict__ Bm, int ldb,
                                             int colBase, int kBase, int lane) {
  const int sel = lane >> 4;
  const int col = colBase + (lane & 15);
  const __bf16* p = Bm + (size_t)col * ldb + kBase + sel * 16;
  v16bf f;
  uint4* fp = reinterpret_cast<uint4*>(&f);
  const uint4* gp = reinterpret_cast<const uint4*>(p);
  fp[0] = gp[0];
  fp[1] = gp[1];
  return f;
}

// ---------------------------------------------------------------------------
// Generic WMMA GEMM:  C[M,N] = A1[M,K1] @ B1[N,K1]^T (+ A2[M,K2] @ B2[N,K2]^T)
//                              (+ bias[N])
// A row-major with stride lda; B row-major (N rows of K); C stride ldc.
// One wave per 32x32 output macro-tile (2x2 WMMA tiles, register reuse:
// 2 A-frag + 2 B-frag loads feed 4 WMMAs per k-step).
// Requires M,N multiples of 32; K multiples of 32.
// ---------------------------------------------------------------------------
__global__ void wmma_gemm_kernel(const __bf16* __restrict__ A1, int lda1, int K1,
                                 const __bf16* __restrict__ B1,
                                 const __bf16* __restrict__ A2, int lda2, int K2,
                                 const __bf16* __restrict__ B2,
                                 const float* __restrict__ bias,
                                 float* __restrict__ C, int ldc, int M, int N) {
  const int lane = threadIdx.x & 31;
  const int wave = blockIdx.x * (blockDim.x >> 5) + (threadIdx.x >> 5);
  const int tilesN = N >> 5;                 // 32-wide macro-tiles
  const int totalTiles = (M >> 5) * tilesN;
  if (wave >= totalTiles) return;
  const int tM = (wave / tilesN) << 5;
  const int tN = (wave % tilesN) << 5;

  v8f acc00 = {0.f, 0.f, 0.f, 0.f, 0.f, 0.f, 0.f, 0.f};
  v8f acc01 = acc00, acc10 = acc00, acc11 = acc00;

  for (int k = 0; k < K1; k += 32) {
    v16bf a0 = load_a_frag(A1, lda1, tM,      k, lane);
    v16bf a1 = load_a_frag(A1, lda1, tM + 16, k, lane);
    v16bf b0 = load_b_frag(B1, K1,   tN,      k, lane);
    v16bf b1 = load_b_frag(B1, K1,   tN + 16, k, lane);
    acc00 = __builtin_amdgcn_wmma_f32_16x16x32_bf16(false, a0, false, b0,
                                                    (short)0, acc00, false, false);
    acc01 = __builtin_amdgcn_wmma_f32_16x16x32_bf16(false, a0, false, b1,
                                                    (short)0, acc01, false, false);
    acc10 = __builtin_amdgcn_wmma_f32_16x16x32_bf16(false, a1, false, b0,
                                                    (short)0, acc10, false, false);
    acc11 = __builtin_amdgcn_wmma_f32_16x16x32_bf16(false, a1, false, b1,
                                                    (short)0, acc11, false, false);
  }
  if (A2 != nullptr) {
    for (int k = 0; k < K2; k += 32) {
      v16bf a0 = load_a_frag(A2, lda2, tM,      k, lane);
      v16bf a1 = load_a_frag(A2, lda2, tM + 16, k, lane);
      v16bf b0 = load_b_frag(B2, K2,   tN,      k, lane);
      v16bf b1 = load_b_frag(B2, K2,   tN + 16, k, lane);
      acc00 = __builtin_amdgcn_wmma_f32_16x16x32_bf16(false, a0, false, b0,
                                                      (short)0, acc00, false, false);
      acc01 = __builtin_amdgcn_wmma_f32_16x16x32_bf16(false, a0, false, b1,
                                                      (short)0, acc01, false, false);
      acc10 = __builtin_amdgcn_wmma_f32_16x16x32_bf16(false, a1, false, b0,
                                                      (short)0, acc10, false, false);
      acc11 = __builtin_amdgcn_wmma_f32_16x16x32_bf16(false, a1, false, b1,
                                                      (short)0, acc11, false, false);
    }
  }

  const int col0 = tN + (lane & 15);
  const int col1 = col0 + 16;
  const float bv0 = bias ? bias[col0] : 0.f;
  const float bv1 = bias ? bias[col1] : 0.f;
  const int rOff = (lane >> 4) * 8;
  for (int r = 0; r < 8; ++r) {
    const size_t row0 = (size_t)(tM + rOff + r) * ldc;
    const size_t row1 = (size_t)(tM + 16 + rOff + r) * ldc;
    C[row0 + col0] = acc00[r] + bv0;
    C[row0 + col1] = acc01[r] + bv1;
    C[row1 + col0] = acc10[r] + bv0;
    C[row1 + col1] = acc11[r] + bv1;
  }
}

// ---------------------------------------------------------------------------
// Elementwise / helper kernels
// ---------------------------------------------------------------------------
__global__ void f2bf_kernel(const float* __restrict__ in, __bf16* __restrict__ out,
                            size_t n) {
  size_t i = (size_t)blockIdx.x * blockDim.x + threadIdx.x;
  size_t stride = (size_t)gridDim.x * blockDim.x;
  for (; i < n; i += stride) out[i] = (__bf16)in[i];
}

// Wq_sum[j,k] = Wq[j,k] + Wq[j,H+k]  (q = [dec,dec]@Wq^T == dec@Wq_sum^T)
// plus fused bias sums for both LSTM layers.
__global__ void prep_small_kernel(const float* __restrict__ Wq, __bf16* __restrict__ WqSum,
                                  const float* __restrict__ bih0, const float* __restrict__ bhh0,
                                  float* __restrict__ b0,
                                  const float* __restrict__ bih1, const float* __restrict__ bhh1,
                                  float* __restrict__ b1) {
  size_t i = (size_t)blockIdx.x * blockDim.x + threadIdx.x;
  size_t stride = (size_t)gridDim.x * blockDim.x;
  const size_t n = (size_t)HH * HH;
  for (size_t j = i; j < n; j += stride) {
    size_t r = j >> 10, c = j & (HH - 1);
    WqSum[j] = (__bf16)(Wq[r * 2 * HH + c] + Wq[r * 2 * HH + HH + c]);
  }
  if (i < GG) {
    b0[i] = bih0[i] + bhh0[i];
    b1[i] = bih1[i] + bhh1[i];
  }
}

__global__ void init_state_kernel(const float* __restrict__ fh, const float* __restrict__ fc,
                                  float* __restrict__ h0, float* __restrict__ c0,
                                  float* __restrict__ h1, float* __restrict__ c1,
                                  __bf16* __restrict__ h0bf, __bf16* __restrict__ h1bf,
                                  float* __restrict__ cov) {
  int i = blockIdx.x * blockDim.x + threadIdx.x;
  if (i < BB * HH) {
    int b = i >> 10, hh = i & (HH - 1);
    float v0 = fh[b * 2 * HH + hh];
    float v1 = fh[b * 2 * HH + HH + hh];
    h0[i] = v0; h1[i] = v1;
    h0bf[i] = (__bf16)v0; h1bf[i] = (__bf16)v1;
    c0[i] = fc[b * 2 * HH + hh];
    c1[i] = fc[b * 2 * HH + HH + hh];
  }
  if (i < BB * SS) cov[i] = 0.f;
}

__device__ __forceinline__ float sigmoidf_(float x) { return 1.f / (1.f + __expf(-x)); }

// LSTM cell pointwise: gates g[B,4H] (i,f,g,o), updates c in place, writes h
// (f32 + bf16, optional second bf16 copy with row stride for the concat buf).
__global__ void lstm_cell_kernel(const float* __restrict__ g, float* __restrict__ c,
                                 float* __restrict__ h, __bf16* __restrict__ hbf,
                                 __bf16* __restrict__ hbf2, int bf2Stride) {
  int i = blockIdx.x * blockDim.x + threadIdx.x;
  if (i >= BB * HH) return;
  int b = i >> 10, hh = i & (HH - 1);
  const float* gb = g + (size_t)b * GG;
  float ig = sigmoidf_(gb[hh]);
  float fg = sigmoidf_(gb[HH + hh]);
  float gv = tanhf(gb[2 * HH + hh]);
  float og = sigmoidf_(gb[3 * HH + hh]);
  float cn = fg * c[i] + ig * gv;
  c[i] = cn;
  float hn = og * tanhf(cn);
  h[i] = hn;
  hbf[i] = (__bf16)hn;
  if (hbf2) hbf2[(size_t)b * bf2Stride + hh] = (__bf16)hn;
}

// e[b,s] = v . tanh(q[b,:] + states_proj[b,s,:] + cov[b,s]*wc + ba)
// one wave per (b,s); coalesced over h; wave shuffle reduction (wave32).
__global__ void attn_e_kernel(const float* __restrict__ q, const float* __restrict__ sp,
                              const float* __restrict__ cov, const float* __restrict__ wc,
                              const float* __restrict__ ba, const float* __restrict__ vv,
                              float* __restrict__ e) {
  int lane = threadIdx.x & 31;
  int gw = blockIdx.x * (blockDim.x >> 5) + (threadIdx.x >> 5);
  if (gw >= BB * SS) return;
  int b = gw >> 9;          // /SS
  int s = gw & (SS - 1);
  float cv = cov[b * SS + s];
  const float* qb = q + (size_t)b * HH;
  const float* spb = sp + ((size_t)b * SS + s) * HH;
  float sum = 0.f;
  for (int h = lane; h < HH; h += 32)
    sum += vv[h] * tanhf(qb[h] + spb[h] + cv * wc[h] + ba[h]);
  for (int off = 16; off; off >>= 1) sum += __shfl_xor(sum, off, 32);
  if (lane == 0) e[b * SS + s] = sum;
}

// softmax over s (block per b, 512 threads); updates cov; writes attn+cov
// into workspace and into the output tensors at timestep t.
__global__ void attn_softmax_kernel(const float* __restrict__ e, float* __restrict__ cov,
                                    float* __restrict__ attn,
                                    float* __restrict__ outAttn, float* __restrict__ outCov,
                                    int t) {
  __shared__ float red[SS];
  int b = blockIdx.x, s = threadIdx.x;
  float ev = e[b * SS + s];
  red[s] = ev;
  __syncthreads();
  for (int off = SS / 2; off; off >>= 1) {
    if (s < off) red[s] = fmaxf(red[s], red[s + off]);
    __syncthreads();
  }
  float mx = red[0];
  __syncthreads();
  float ex = __expf(ev - mx);
  red[s] = ex;
  __syncthreads();
  for (int off = SS / 2; off; off >>= 1) {
    if (s < off) red[s] += red[s + off];
    __syncthreads();
  }
  float a = ex / red[0];
  float cv = cov[b * SS + s] + a;
  attn[b * SS + s] = a;
  cov[b * SS + s] = cv;
  size_t o = ((size_t)b * TT + t) * SS + s;
  outAttn[o] = a;
  outCov[o] = cv;
}

// ctx[b,h] = sum_s attn[b,s] * states[b,s,h]; block per b, 1024 threads.
__global__ void ctx_kernel(const float* __restrict__ attn, const float* __restrict__ states,
                           float* __restrict__ ctx, __bf16* __restrict__ catbf) {
  __shared__ float sa[SS];
  int b = blockIdx.x, h = threadIdx.x;
  if (h < SS) sa[h] = attn[b * SS + h];
  __syncthreads();
  const float* st = states + (size_t)b * SS * HH + h;
  float acc = 0.f;
  for (int s = 0; s < SS; ++s) acc += sa[s] * st[(size_t)s * HH];
  ctx[(size_t)b * HH + h] = acc;
  catbf[(size_t)b * 2 * HH + HH + h] = (__bf16)acc;   // [dec | ctx]
}

// p_gen[b] = sigmoid([ctx,dec,emb] . Wp + bp); one wave per b.
__global__ void pgen_kernel(const float* __restrict__ ctx, const float* __restrict__ dec,
                            const float* __restrict__ emb, const float* __restrict__ Wp,
                            const float* __restrict__ bp, float* __restrict__ outP, int t) {
  int b = blockIdx.x, lane = threadIdx.x;
  const float* eb = emb + ((size_t)b * TT + t) * INW;
  float sum = 0.f;
  for (int i = lane; i < HH; i += 32) sum += ctx[(size_t)b * HH + i] * Wp[i];
  for (int i = lane; i < HH; i += 32) sum += dec[(size_t)b * HH + i] * Wp[HH + i];
  for (int i = lane; i < INW; i += 32) sum += eb[i] * Wp[2 * HH + i];
  for (int off = 16; off; off >>= 1) sum += __shfl_xor(sum, off, 32);
  if (lane == 0) outP[(size_t)b * TT + t] = sigmoidf_(sum + bp[0]);
}

// ---------------------------------------------------------------------------
// Host orchestration
// ---------------------------------------------------------------------------
static inline size_t align256(size_t x) { return (x + 255) & ~(size_t)255; }

static inline void launch_gemm(const __bf16* A1, int lda1, int K1, const __bf16* B1,
                               const __bf16* A2, int lda2, int K2, const __bf16* B2,
                               const float* bias, float* C, int ldc, int M, int N,
                               hipStream_t stream) {
  int tiles = (M >> 5) * (N >> 5);   // 32x32 macro-tiles
  int blocks = (tiles + 3) >> 2;     // 4 waves (128 threads) per block
  wmma_gemm_kernel<<<blocks, 128, 0, stream>>>(A1, lda1, K1, B1, A2, lda2, K2, B2,
                                               bias, C, ldc, M, N);
}

static inline void launch_f2bf(const float* in, __bf16* out, size_t n, hipStream_t stream) {
  int blocks = (int)((n + 2047) / 2048);
  if (blocks > 4096) blocks = 4096;
  f2bf_kernel<<<blocks, 256, 0, stream>>>(in, out, n);
}

extern "C" void kernel_launch(void* const* d_in, const int* in_sizes, int n_in,
                              void* d_out, int out_size, void* d_ws, size_t ws_size,
                              hipStream_t stream) {
  const float* emb    = (const float*)d_in[0];   // (B,T,IN)
  const float* fh     = (const float*)d_in[1];   // (B,2,H)
  const float* fc     = (const float*)d_in[2];   // (B,2,H)
  const float* states = (const float*)d_in[3];   // (B,S,H)
  const float* Wih0   = (const float*)d_in[4];
  const float* Whh0   = (const float*)d_in[5];
  const float* bih0   = (const float*)d_in[6];
  const float* bhh0   = (const float*)d_in[7];
  const float* Wih1   = (const float*)d_in[8];
  const float* Whh1   = (const float*)d_in[9];
  const float* bih1   = (const float*)d_in[10];
  const float* bhh1   = (const float*)d_in[11];
  const float* Wq     = (const float*)d_in[12];  // (H,2H)
  const float* Ws     = (const float*)d_in[13];  // (H,H)
  const float* wc     = (const float*)d_in[14];
  const float* ba     = (const float*)d_in[15];
  const float* vv     = (const float*)d_in[16];
  const float* Wc1    = (const float*)d_in[17];  // (H,2H)
  const float* bc1    = (const float*)d_in[18];
  const float* Wc2    = (const float*)d_in[19];  // (H,H)
  const float* bc2    = (const float*)d_in[20];
  const float* Wp     = (const float*)d_in[21];  // (1,2H+IN)
  const float* bp     = (const float*)d_in[22];

  float* outDec  = (float*)d_out;                                  // (B,T,H)
  float* outP    = outDec + (size_t)BB * TT * HH;                  // (B,T,1)
  float* outAttn = outP + (size_t)BB * TT;                         // (B,T,S)
  float* outCov  = outAttn + (size_t)BB * TT * SS;                 // (B,T,S)

  // --- workspace bump allocator (256B aligned) ---
  char* base = (char*)d_ws;
  size_t off = 0;
  auto alloc = [&](size_t bytes) -> void* {
    void* p = base + off;
    off = align256(off + bytes);
    return p;
  };
  float*  sp       = (float*) alloc((size_t)BB * SS * HH * 4);   // states_proj
  __bf16* statesBf = (__bf16*)alloc((size_t)BB * SS * HH * 2);
  __bf16* embBf    = (__bf16*)alloc((size_t)BB * TT * INW * 2);
  __bf16* Wih0Bf   = (__bf16*)alloc((size_t)GG * INW * 2);
  __bf16* Whh0Bf   = (__bf16*)alloc((size_t)GG * HH * 2);
  __bf16* Wih1Bf   = (__bf16*)alloc((size_t)GG * HH * 2);
  __bf16* Whh1Bf   = (__bf16*)alloc((size_t)GG * HH * 2);
  __bf16* WqSumBf  = (__bf16*)alloc((size_t)HH * HH * 2);
  __bf16* Wc1Bf    = (__bf16*)alloc((size_t)HH * 2 * HH * 2);
  __bf16* Wc2Bf    = (__bf16*)alloc((size_t)HH * HH * 2);
  __bf16* WsBf     = (__bf16*)alloc((size_t)HH * HH * 2);
  float*  b0sum    = (float*) alloc((size_t)GG * 4);
  float*  b1sum    = (float*) alloc((size_t)GG * 4);
  float*  h0       = (float*) alloc((size_t)BB * HH * 4);
  float*  c0       = (float*) alloc((size_t)BB * HH * 4);
  float*  h1       = (float*) alloc((size_t)BB * HH * 4);
  float*  c1       = (float*) alloc((size_t)BB * HH * 4);
  __bf16* h0bf     = (__bf16*)alloc((size_t)BB * HH * 2);
  __bf16* h1bf     = (__bf16*)alloc((size_t)BB * HH * 2);
  float*  g0       = (float*) alloc((size_t)BB * GG * 4);
  float*  g1       = (float*) alloc((size_t)BB * GG * 4);
  float*  q        = (float*) alloc((size_t)BB * HH * 4);
  float*  e        = (float*) alloc((size_t)BB * SS * 4);
  float*  attn     = (float*) alloc((size_t)BB * SS * 4);
  float*  cov      = (float*) alloc((size_t)BB * SS * 4);
  float*  ctx      = (float*) alloc((size_t)BB * HH * 4);
  __bf16* catBf    = (__bf16*)alloc((size_t)BB * 2 * HH * 2);    // [dec|ctx]
  float*  tmp1     = (float*) alloc((size_t)BB * HH * 4);
  __bf16* tmp1Bf   = (__bf16*)alloc((size_t)BB * HH * 2);
  (void)ws_size; (void)in_sizes; (void)n_in; (void)out_size;

  // --- one-time prep: bf16 conversions, fused weights/biases, states_proj ---
  launch_f2bf(states, statesBf, (size_t)BB * SS * HH, stream);
  launch_f2bf(emb, embBf, (size_t)BB * TT * INW, stream);
  launch_f2bf(Wih0, Wih0Bf, (size_t)GG * INW, stream);
  launch_f2bf(Whh0, Whh0Bf, (size_t)GG * HH, stream);
  launch_f2bf(Wih1, Wih1Bf, (size_t)GG * HH, stream);
  launch_f2bf(Whh1, Whh1Bf, (size_t)GG * HH, stream);
  launch_f2bf(Wc1, Wc1Bf, (size_t)HH * 2 * HH, stream);
  launch_f2bf(Wc2, Wc2Bf, (size_t)HH * HH, stream);
  launch_f2bf(Ws, WsBf, (size_t)HH * HH, stream);
  prep_small_kernel<<<2048, 256, 0, stream>>>(Wq, WqSumBf, bih0, bhh0, b0sum,
                                              bih1, bhh1, b1sum);
  init_state_kernel<<<(BB * HH + 255) / 256, 256, 0, stream>>>(
      fh, fc, h0, c0, h1, c1, h0bf, h1bf, cov);
  // states_proj = states @ Ws^T : (B*S, H) x (H, H)
  launch_gemm(statesBf, HH, HH, WsBf, nullptr, 0, 0, nullptr, nullptr,
              sp, HH, BB * SS, HH, stream);

  // --- 64 sequential decode steps ---
  for (int t = 0; t < TT; ++t) {
    // LSTM layer 0 gates: emb_t @ Wih0^T + h0 @ Whh0^T + b
    launch_gemm(embBf + (size_t)t * INW, TT * INW, INW, Wih0Bf,
                h0bf, HH, HH, Whh0Bf, b0sum, g0, GG, BB, GG, stream);
    lstm_cell_kernel<<<(BB * HH + 255) / 256, 256, 0, stream>>>(
        g0, c0, h0, h0bf, nullptr, 0);
    // LSTM layer 1 gates: h0 @ Wih1^T + h1 @ Whh1^T + b
    launch_gemm(h0bf, HH, HH, Wih1Bf, h1bf, HH, HH, Whh1Bf,
                b1sum, g1, GG, BB, GG, stream);
    lstm_cell_kernel<<<(BB * HH + 255) / 256, 256, 0, stream>>>(
        g1, c1, h1, h1bf, catBf, 2 * HH);          // dec -> cat[:, 0:H]
    // q = dec @ (Wq[:, :H] + Wq[:, H:])^T
    launch_gemm(h1bf, HH, HH, WqSumBf, nullptr, 0, 0, nullptr, nullptr,
                q, HH, BB, HH, stream);
    // attention scores, softmax (+cov update), context
    attn_e_kernel<<<(BB * SS) / 8, 256, 0, stream>>>(q, sp, cov, wc, ba, vv, e);
    attn_softmax_kernel<<<BB, SS, 0, stream>>>(e, cov, attn, outAttn, outCov, t);
    ctx_kernel<<<BB, HH, 0, stream>>>(attn, states, ctx, catBf);
    // p_gen
    pgen_kernel<<<BB, 32, 0, stream>>>(ctx, h1, emb, Wp, bp, outP, t);
    // output head: ([dec|ctx] @ Wc1^T + bc1) @ Wc2^T + bc2
    launch_gemm(catBf, 2 * HH, 2 * HH, Wc1Bf, nullptr, 0, 0, nullptr,
                bc1, tmp1, HH, BB, HH, stream);
    launch_f2bf(tmp1, tmp1Bf, (size_t)BB * HH, stream);
    launch_gemm(tmp1Bf, HH, HH, Wc2Bf, nullptr, 0, 0, nullptr,
                bc2, outDec + (size_t)t * HH, TT * HH, BB, HH, stream);
  }
}
